// GroupedQueryAttention_85882166051171
// MI455X (gfx1250) — matrix operand order repaired and checked by reading, not compiled, not run on hardware
//
#include <hip/hip_runtime.h>
#include <hip/hip_bf16.h>
#include <stdint.h>

// Problem constants (match reference)
#define B_SZ    2
#define S_LEN   2048
#define D_MODEL 2048
#define N_HEADS 32
#define N_KV    8
#define HDIM    64
// N_KV_GROUPS = 4, scale = 1/sqrt(64) = 0.125

typedef __attribute__((ext_vector_type(16))) __bf16 bf16x16;
typedef __attribute__((ext_vector_type(8)))  float  f32x8;
typedef __attribute__((ext_vector_type(4)))  unsigned int v4u;
typedef __attribute__((ext_vector_type(8)))  int          v8i;
typedef __attribute__((ext_vector_type(4)))  int          v4i;

union B16Frag { bf16x16 v; uint4 q[2]; };

__device__ __forceinline__ unsigned short f2bf(float f) {
  union { float f; unsigned int u; } cv; cv.f = f;
  unsigned int u = cv.u;
  u += 0x7FFFu + ((u >> 16) & 1u);     // round-to-nearest-even
  return (unsigned short)(u >> 16);
}

// ---------------------------------------------------------------------------
// Tensor Data Mover: DMA a 2D bf16 tile [tile_n rows x tile_k cols] from a
// row-major matrix (row stride = K elements) into LDS. D# per ISA ch.8:
//   group0: [1:0]=count=1, [63:32]=lds_addr, [120:64]=global_addr, [127:126]=type 2
//   group1: [17:16]=data_size(1=2B), [79:48]=tensor_dim0, [111:80]=tensor_dim1,
//           [127:112]=tile_dim0, [143:128]=tile_dim1, [207:160]=tensor_dim0_stride
// This toolchain exposes the 6-arg builtin: (v4u, v8i, v4i, v4i, v8i, i32 cpol).
__device__ __forceinline__ void tdm_load_2d_bf16(unsigned lds_off, const void* gptr,
                                                 int K_elems, int N_rows,
                                                 int tile_k, int tile_n) {
  unsigned long long ga = (unsigned long long)gptr;
  v4u g0;
  g0.x = 1u;                                            // count=1 (valid), user mode
  g0.y = lds_off;                                       // lds_addr (bytes)
  g0.z = (unsigned)(ga & 0xFFFFFFFFu);                  // global_addr[95:64]
  g0.w = (unsigned)((ga >> 32) & 0x01FFFFFFu) | (2u << 30);  // addr[56:32] | type=2
  v8i g1;
  g1[0] = 0x00010000;                                   // data_size=1 -> 2 bytes
  g1[1] = (int)(((unsigned)K_elems & 0xFFFFu) << 16);   // tensor_dim0[15:0]
  g1[2] = (int)((((unsigned)K_elems >> 16) & 0xFFFFu) | (((unsigned)N_rows & 0xFFFFu) << 16));
  g1[3] = (int)((((unsigned)N_rows >> 16) & 0xFFFFu) | (((unsigned)tile_k & 0xFFFFu) << 16));
  g1[4] = (int)((unsigned)tile_n & 0xFFFFu);            // tile_dim1 (tile_dim2=0)
  g1[5] = K_elems;                                      // tensor_dim0_stride[31:0]
  g1[6] = 0;                                            // stride hi, dim1_stride lo
  g1[7] = 0;
  v4i z4; z4[0] = 0; z4[1] = 0; z4[2] = 0; z4[3] = 0;   // groups 2/3 unused (2D)
  v8i z8; z8[0] = 0; z8[1] = 0; z8[2] = 0; z8[3] = 0;
  z8[4] = 0; z8[5] = 0; z8[6] = 0; z8[7] = 0;
  __builtin_amdgcn_tensor_load_to_lds(g0, g1, z4, z4, z8, 0);
}

// ---------------------------------------------------------------------------
// Elementwise fp32 -> bf16
__global__ void __launch_bounds__(256) cvt_f32_bf16(const float* __restrict__ in,
                                                    unsigned short* __restrict__ out, int n) {
  int i = blockIdx.x * 256 + threadIdx.x;
  if (i < n) out[i] = f2bf(in[i]);
}

// fp32 [K,N] -> bf16 [N,K]  (weights pre-transposed so WMMA B-fragments load contiguously)
__global__ void __launch_bounds__(256) transpose_cvt(const float* __restrict__ in,
                                                     unsigned short* __restrict__ out,
                                                     int K, int N) {
  int i = blockIdx.x * 256 + threadIdx.x;
  if (i >= K * N) return;
  int k = i / N, n = i - k * N;
  out[(size_t)n * K + k] = f2bf(in[i]);
}

// ---------------------------------------------------------------------------
// bf16 WMMA GEMM with TDM-staged weight panels:
//   C[M,N] (f32) = A[M,K] (bf16 row-major) * Bt[N,K]^T
// Block = 256 threads = 8 waves; block tile 128(M) x 64(N); K chunk = 64.
// Wave 0 DMAs the 64x64 bf16 Bt panel into LDS via tensor_load_to_lds
// (double-buffered, TENSORcnt), all 8 waves consume it from LDS -> 8x less
// global weight traffic and DMA/compute overlap. A-fragments stream from
// global per wave.
// A-fragment (16x32 bf16, ISA 7.12.2): lanes 0-15 row m: K0-7 @ +0, K16-23 @ +32B;
//                                      lanes 16-31:      K8-15 @ +16, K24-31 @ +48B.
// B-fragment (32x16 bf16): lane n = column n; lanes 0-15 hold K0-7, K16-23 (+0,+32),
//                          lanes 16-31 hold K8-15, K24-31 (+16,+48).
__global__ void __launch_bounds__(256) gemm_bf16_wmma(const unsigned short* __restrict__ A,
                                                      const unsigned short* __restrict__ Bt,
                                                      float* __restrict__ C,
                                                      int M, int N, int K) {
  __shared__ __align__(16) unsigned short bpanel[2][64 * 64];   // 2 x 8KB

  const int wave = threadIdx.x >> 5;
  const int lane = threadIdx.x & 31;
  const int hl   = lane >> 4;       // half-wave select
  const int l15  = lane & 15;

  const int m0 = blockIdx.y * 128 + wave * 16;
  const int n0 = blockIdx.x * 64;

  f32x8 acc[4];
  for (int t = 0; t < 4; ++t) acc[t] = (f32x8){0.f,0.f,0.f,0.f,0.f,0.f,0.f,0.f};

  const char* arow = (const char*)(A + (size_t)(m0 + l15) * K);
  const unsigned short* bt0 = Bt + (size_t)n0 * K;

  const int iters = K >> 6;                       // K / 64
  if (wave == 0) {
    tdm_load_2d_bf16((unsigned)(size_t)(const void*)bpanel[0], bt0, K, N - n0, 64, 64);
  }

  for (int i = 0; i < iters; ++i) {
    const int k = i << 6;
    if (wave == 0) {
      __builtin_amdgcn_s_wait_tensorcnt(0);       // current panel resident
      if (i + 1 < iters)                          // overlap next DMA with compute
        tdm_load_2d_bf16((unsigned)(size_t)(const void*)bpanel[(i + 1) & 1],
                         bt0 + (k + 64), K, N - n0, 64, 64);
    }
    __syncthreads();

    const unsigned short* bp = bpanel[i & 1];
    if (k + 128 < K) __builtin_prefetch(arow + (size_t)(k + 128) * 2, 0, 1);

    for (int kk = 0; kk < 64; kk += 32) {
      B16Frag a;
      const char* ap = arow + (size_t)(k + kk) * 2;
      a.q[0] = *(const uint4*)(ap + hl * 16);
      a.q[1] = *(const uint4*)(ap + 32 + hl * 16);
      for (int t = 0; t < 4; ++t) {
        const char* bpp = (const char*)(bp + (size_t)(t * 16 + l15) * 64 + kk);
        B16Frag b;
        b.q[0] = *(const uint4*)(bpp + hl * 16);
        b.q[1] = *(const uint4*)(bpp + 32 + hl * 16);
        acc[t] = __builtin_amdgcn_wmma_f32_16x16x32_bf16(false, a.v, false, b.v,
                                                         (short)0, acc[t], false, false);
      }
    }
    __syncthreads();                              // panel fully consumed before overwrite
  }

  // C layout: VGPR v -> row (m0 + v + 8*half), col = n0 + t*16 + l15
  for (int v = 0; v < 8; ++v) {
    float* crow = C + (size_t)(m0 + v + hl * 8) * N + n0;
    for (int t = 0; t < 4; ++t) crow[t * 16 + l15] = acc[t][v];
  }
}

// ---------------------------------------------------------------------------
// RoPE for Q/K: in fp32 [B,S,Hn,64] -> out bf16 [B,Hn,S,64]
__global__ void __launch_bounds__(256) rope_qk(const float* __restrict__ in,
                                               const float* __restrict__ cosf,
                                               const float* __restrict__ sinf,
                                               unsigned short* __restrict__ out, int Hn) {
  int i = blockIdx.x * 256 + threadIdx.x;            // over B*S*Hn*32 pairs
  int total = B_SZ * S_LEN * Hn * 32;
  if (i >= total) return;
  int p = i & 31; int t = i >> 5;
  int hh = t % Hn; t /= Hn;
  int s = t % S_LEN; int b = t / S_LEN;
  const float* src = in + (((size_t)(b * S_LEN + s) * Hn + hh) * HDIM) + 2 * p;
  float xe = src[0], xo = src[1];
  float c = cosf[s * 32 + p], sn = sinf[s * 32 + p];
  unsigned short* dst = out + (((size_t)(b * Hn + hh) * S_LEN + s) * HDIM) + 2 * p;
  dst[0] = f2bf(xe * c - xo * sn);
  dst[1] = f2bf(xe * sn + xo * c);
}

// V: fp32 [B,S,8,64] -> bf16 transposed [B,8,64,S] (so PV B-fragments are contiguous)
__global__ void __launch_bounds__(256) cvt_v_transpose(const float* __restrict__ in,
                                                       unsigned short* __restrict__ out) {
  int i = blockIdx.x * 256 + threadIdx.x;            // over B*S*8*64
  if (i >= B_SZ * S_LEN * N_KV * HDIM) return;
  int d = i & 63; int t = i >> 6;
  int kv = t & 7; t >>= 3;
  int s = t & (S_LEN - 1); int b = t >> 11;
  out[(((size_t)(b * N_KV + kv) * HDIM + d) * S_LEN) + s] = f2bf(in[i]);
}

// ---------------------------------------------------------------------------
// Flash attention (causal). One wave owns a 16-row query tile; K loop in 32-key
// chunks; scores via 2x2 WMMA, online softmax with width-16 shuffles, P routed
// C-layout -> A-layout through per-wave LDS, PV via 4 WMMA per chunk.
__global__ void __launch_bounds__(256) flash_attn_wmma(const unsigned short* __restrict__ qb,  // [B,32,S,64]
                                                       const unsigned short* __restrict__ kb,  // [B,8,S,64]
                                                       const unsigned short* __restrict__ vtb, // [B,8,64,S]
                                                       unsigned short* __restrict__ ob) {      // [B,S,2048]
  __shared__ __align__(16) unsigned short lds_p[8][16 * 32];

  const int wave = threadIdx.x >> 5;
  const int lane = threadIdx.x & 31;
  const int hl   = lane >> 4;
  const int l15  = lane & 15;

  const int gw   = blockIdx.x * 8 + wave;      // 0 .. B*H*(S/16)-1
  const int tile = gw & 127;                   // S/16 = 128 tiles
  const int bh   = gw >> 7;
  const int h    = bh & 31;
  const int b    = bh >> 5;
  const int kv   = h >> 2;                     // 4 Q heads per KV head
  const int q0   = tile * 16;

  // Q A-fragments (16x64 -> two 16x32 frags)
  const char* qrow = (const char*)(qb + (((size_t)(b * N_HEADS + h) * S_LEN) + (q0 + l15)) * HDIM);
  B16Frag aq0, aq1;
  aq0.q[0] = *(const uint4*)(qrow + hl * 16);
  aq0.q[1] = *(const uint4*)(qrow + 32 + hl * 16);
  aq1.q[0] = *(const uint4*)(qrow + 64 + hl * 16);
  aq1.q[1] = *(const uint4*)(qrow + 96 + hl * 16);

  float m_run[8], l_run[8];
  f32x8 o[4];
  for (int v = 0; v < 8; ++v) { m_run[v] = -1e30f; l_run[v] = 0.f; }
  for (int t = 0; t < 4; ++t) o[t] = (f32x8){0.f,0.f,0.f,0.f,0.f,0.f,0.f,0.f};

  const unsigned short* kbase = kb  + (size_t)(b * N_KV + kv) * S_LEN * HDIM;
  const unsigned short* vbase = vtb + (size_t)(b * N_KV + kv) * HDIM * S_LEN;
  unsigned short* lp = lds_p[wave];

  for (int j0 = 0; j0 <= q0 + 15; j0 += 32) {
    // S = Q @ K^T for 32 keys -> two 16x16 f32 tiles
    f32x8 c01[2];
    for (int t = 0; t < 2; ++t) {
      const char* kr = (const char*)(kbase + (size_t)(j0 + t * 16 + l15) * HDIM);
      B16Frag bk0, bk1;
      bk0.q[0] = *(const uint4*)(kr + hl * 16);
      bk0.q[1] = *(const uint4*)(kr + 32 + hl * 16);
      bk1.q[0] = *(const uint4*)(kr + 64 + hl * 16);
      bk1.q[1] = *(const uint4*)(kr + 96 + hl * 16);
      f32x8 c = (f32x8){0.f,0.f,0.f,0.f,0.f,0.f,0.f,0.f};
      c = __builtin_amdgcn_wmma_f32_16x16x32_bf16(false, aq0.v, false, bk0.v, (short)0, c, false, false);
      c = __builtin_amdgcn_wmma_f32_16x16x32_bf16(false, aq1.v, false, bk1.v, (short)0, c, false, false);
      c01[t] = c;
    }

    // online softmax: per C-VGPR v, row = q0 + v + 8*half, col = lane&15 (+16 for tile 1)
    float p0[8], p1[8], alpha[8];
    for (int v = 0; v < 8; ++v) {
      int qrow_i = q0 + v + hl * 8;
      float s0 = c01[0][v] * 0.125f + ((j0 + l15)      <= qrow_i ? 0.f : -1e9f);
      float s1 = c01[1][v] * 0.125f + ((j0 + 16 + l15) <= qrow_i ? 0.f : -1e9f);
      float mx = fmaxf(s0, s1);
      mx = fmaxf(mx, __shfl_xor(mx, 1, 16));
      mx = fmaxf(mx, __shfl_xor(mx, 2, 16));
      mx = fmaxf(mx, __shfl_xor(mx, 4, 16));
      mx = fmaxf(mx, __shfl_xor(mx, 8, 16));
      float mnew = fmaxf(m_run[v], mx);
      alpha[v] = __expf(m_run[v] - mnew);
      p0[v] = __expf(s0 - mnew);
      p1[v] = __expf(s1 - mnew);
      float rs = p0[v] + p1[v];
      rs += __shfl_xor(rs, 1, 16);
      rs += __shfl_xor(rs, 2, 16);
      rs += __shfl_xor(rs, 4, 16);
      rs += __shfl_xor(rs, 8, 16);
      l_run[v] = l_run[v] * alpha[v] + rs;
      m_run[v] = mnew;
    }
    for (int t = 0; t < 4; ++t)
      for (int v = 0; v < 8; ++v) o[t][v] *= alpha[v];

    // P (C layout) -> LDS (row-major 16x32 bf16) -> A-fragment
    for (int v = 0; v < 8; ++v) {
      int r = v + hl * 8;
      lp[r * 32 + l15]      = f2bf(p0[v]);
      lp[r * 32 + 16 + l15] = f2bf(p1[v]);
    }
    __builtin_amdgcn_fence(__ATOMIC_RELEASE, "wavefront");
    __builtin_amdgcn_wave_barrier();              // wave-internal: LDS is in-order per wave
    __builtin_amdgcn_fence(__ATOMIC_ACQUIRE, "wavefront");

    B16Frag ap;
    const char* pp = (const char*)(lp + l15 * 32);
    ap.q[0] = *(const uint4*)(pp + hl * 16);
    ap.q[1] = *(const uint4*)(pp + 32 + hl * 16);

    // O += P @ V  (V^T rows are contiguous over key positions)
    for (int t = 0; t < 4; ++t) {
      const char* vr = (const char*)(vbase + (size_t)(t * 16 + l15) * S_LEN + j0);
      B16Frag bv;
      bv.q[0] = *(const uint4*)(vr + hl * 16);
      bv.q[1] = *(const uint4*)(vr + 32 + hl * 16);
      o[t] = __builtin_amdgcn_wmma_f32_16x16x32_bf16(false, ap.v, false, bv.v, (short)0, o[t], false, false);
    }
  }

  // normalize + store bf16 to [B,S,H*64]
  for (int v = 0; v < 8; ++v) {
    float inv = 1.f / l_run[v];
    int srow = q0 + v + hl * 8;
    unsigned short* orow = ob + ((size_t)b * S_LEN + srow) * (N_HEADS * HDIM) + h * HDIM;
    for (int t = 0; t < 4; ++t) orow[t * 16 + l15] = f2bf(o[t][v] * inv);
  }
}

// ---------------------------------------------------------------------------
extern "C" void kernel_launch(void* const* d_in, const int* in_sizes, int n_in,
                              void* d_out, int out_size, void* d_ws, size_t ws_size,
                              hipStream_t stream) {
  const float* x    = (const float*)d_in[0];
  const float* fcos = (const float*)d_in[1];
  const float* fsin = (const float*)d_in[2];
  // d_in[3] = mask (unused; causal mask computed inline)
  const float* wq = (const float*)d_in[4];
  const float* wk = (const float*)d_in[5];
  const float* wv = (const float*)d_in[6];
  const float* wo = (const float*)d_in[7];

  const size_t BS = (size_t)B_SZ * S_LEN;           // 4096
  const size_t DM = D_MODEL;                        // 2048
  const size_t HD = (size_t)N_HEADS * HDIM;         // 2048
  const size_t KVD = (size_t)N_KV * HDIM;           // 512

  char* w = (char*)d_ws;
  size_t off = 0;
  auto alloc = [&](size_t bytes) -> void* {
    void* p = w + off; off = (off + bytes + 255) & ~(size_t)255; return p;
  };
  unsigned short* xb  = (unsigned short*)alloc(BS * DM * 2);
  unsigned short* wqT = (unsigned short*)alloc(HD * DM * 2);    // [2048 N][2048 K]
  unsigned short* wkT = (unsigned short*)alloc(KVD * DM * 2);   // [512][2048]
  unsigned short* wvT = (unsigned short*)alloc(KVD * DM * 2);
  unsigned short* woT = (unsigned short*)alloc(DM * HD * 2);    // [2048][2048]
  float*          qf  = (float*)alloc(BS * HD * 4);
  float*          kf  = (float*)alloc(BS * KVD * 4);
  float*          vf  = (float*)alloc(BS * KVD * 4);
  unsigned short* qb  = (unsigned short*)alloc(BS * HD * 2);    // [B,32,S,64]
  unsigned short* kb  = (unsigned short*)alloc(BS * KVD * 2);   // [B,8,S,64]
  unsigned short* vtb = (unsigned short*)alloc(BS * KVD * 2);   // [B,8,64,S]
  unsigned short* ob  = (unsigned short*)alloc(BS * HD * 2);    // [B,S,2048]

  // 1) downconvert activations + transpose/downconvert weights
  cvt_f32_bf16<<<(int)(BS * DM / 256), 256, 0, stream>>>(x, xb, (int)(BS * DM));
  transpose_cvt<<<(int)(DM * HD  / 256), 256, 0, stream>>>(wq, wqT, (int)DM, (int)HD);
  transpose_cvt<<<(int)(DM * KVD / 256), 256, 0, stream>>>(wk, wkT, (int)DM, (int)KVD);
  transpose_cvt<<<(int)(DM * KVD / 256), 256, 0, stream>>>(wv, wvT, (int)DM, (int)KVD);
  transpose_cvt<<<(int)(HD * DM  / 256), 256, 0, stream>>>(wo, woT, (int)HD, (int)DM);

  // 2) QKV projections (WMMA GEMMs, TDM-staged weight panels)
  gemm_bf16_wmma<<<dim3((unsigned)(HD / 64),  (unsigned)(BS / 128)), 256, 0, stream>>>(xb, wqT, qf, (int)BS, (int)HD,  (int)DM);
  gemm_bf16_wmma<<<dim3((unsigned)(KVD / 64), (unsigned)(BS / 128)), 256, 0, stream>>>(xb, wkT, kf, (int)BS, (int)KVD, (int)DM);
  gemm_bf16_wmma<<<dim3((unsigned)(KVD / 64), (unsigned)(BS / 128)), 256, 0, stream>>>(xb, wvT, vf, (int)BS, (int)KVD, (int)DM);

  // 3) RoPE + layout changes (to WMMA-fragment-friendly layouts)
  rope_qk<<<(int)(BS * N_HEADS * 32 / 256), 256, 0, stream>>>(qf, fcos, fsin, qb, N_HEADS);
  rope_qk<<<(int)(BS * N_KV * 32 / 256),    256, 0, stream>>>(kf, fcos, fsin, kb, N_KV);
  cvt_v_transpose<<<(int)(BS * KVD / 256),  256, 0, stream>>>(vf, vtb);

  // 4) flash attention: B*H*(S/16) = 8192 waves, 8 waves/block
  flash_attn_wmma<<<1024, 256, 0, stream>>>(qb, kb, vtb, ob);

  // 5) output projection -> fp32 d_out
  gemm_bf16_wmma<<<dim3((unsigned)(DM / 64), (unsigned)(BS / 128)), 256, 0, stream>>>(ob, woT, (float*)d_out, (int)BS, (int)DM, (int)HD);
}
